// ConvEmbedding_33157147525315
// MI455X (gfx1250) — matrix operand order repaired and checked by reading, not compiled, not run on hardware
//
#include <hip/hip_runtime.h>
#include <hip/hip_bf16.h>

typedef __attribute__((ext_vector_type(16))) _Float16 v16h;
typedef __attribute__((ext_vector_type(8)))  _Float16 v8h;
typedef __attribute__((ext_vector_type(8)))  float    v8f;

#define NROWS 8192
#define TOPK  15

// ---------------------------------------------------------------------------
// Stage 1: conv1 (1->8, k=7, L 25->19) + ReLU, accumulate BN stats for 8 ch.
// stats layout: s1[0..7]=sum, s1[8..15]=sumsq
// ---------------------------------------------------------------------------
__global__ __launch_bounds__(256) void conv1_kernel(const float* __restrict__ x,
                                                    const float* __restrict__ w,
                                                    const float* __restrict__ b,
                                                    float* __restrict__ h1,
                                                    float* __restrict__ s1) {
    __shared__ float ls[8], lss[8];
    if (threadIdx.x < 8) { ls[threadIdx.x] = 0.f; lss[threadIdx.x] = 0.f; }
    __syncthreads();
    int id = blockIdx.x * 256 + threadIdx.x;
    const int total = NROWS * 8 * 19;
    if (id < total) {
        int n   = id / (8 * 19);
        int rem = id % (8 * 19);
        int c   = rem / 19;
        int t   = rem % 19;
        float acc = b[c];
        const float* xr = x + n * 25;
        const float* wr = w + c * 7;
        #pragma unroll
        for (int k = 0; k < 7; ++k) acc += xr[t + k] * wr[k];
        float val = acc > 0.f ? acc : 0.f;
        h1[id] = val;                     // [n][c][t] == id layout
        atomicAdd(&ls[c], val);
        atomicAdd(&lss[c], val * val);
    }
    __syncthreads();
    if (threadIdx.x < 8) {
        atomicAdd(&s1[threadIdx.x],     ls[threadIdx.x]);
        atomicAdd(&s1[8 + threadIdx.x], lss[threadIdx.x]);
    }
}

// ---------------------------------------------------------------------------
// Stage 2: BN1 (folded) + conv2 (8->16, k=5, 19->15) + ReLU + stats (16 ch)
// ---------------------------------------------------------------------------
__global__ __launch_bounds__(256) void conv2_kernel(const float* __restrict__ h1,
                                                    const float* __restrict__ s1,
                                                    const float* __restrict__ g1,
                                                    const float* __restrict__ bb1,
                                                    const float* __restrict__ w,
                                                    const float* __restrict__ b,
                                                    float* __restrict__ h2,
                                                    float* __restrict__ s2) {
    __shared__ float scale[8], shift[8];
    __shared__ float ls[16], lss[16];
    if (threadIdx.x < 8) {
        const float cnt = (float)NROWS * 19.f;
        float m = s1[threadIdx.x] / cnt;
        float v = s1[8 + threadIdx.x] / cnt - m * m;
        float inv = rsqrtf(v + 1e-5f) * g1[threadIdx.x];
        scale[threadIdx.x] = inv;
        shift[threadIdx.x] = bb1[threadIdx.x] - m * inv;
    }
    if (threadIdx.x < 16) { ls[threadIdx.x] = 0.f; lss[threadIdx.x] = 0.f; }
    __syncthreads();
    int id = blockIdx.x * 256 + threadIdx.x;
    const int total = NROWS * 16 * 15;
    if (id < total) {
        int n   = id / (16 * 15);
        int rem = id % (16 * 15);
        int c   = rem / 15;
        int t   = rem % 15;
        float acc = b[c];
        #pragma unroll
        for (int ci = 0; ci < 8; ++ci) {
            float sc = scale[ci], sh = shift[ci];
            const float* hr = h1 + n * 152 + ci * 19;
            const float* wr = w + c * 40 + ci * 5;
            #pragma unroll
            for (int k = 0; k < 5; ++k) acc += (hr[t + k] * sc + sh) * wr[k];
        }
        float val = acc > 0.f ? acc : 0.f;
        h2[id] = val;
        atomicAdd(&ls[c], val);
        atomicAdd(&lss[c], val * val);
    }
    __syncthreads();
    if (threadIdx.x < 16) {
        atomicAdd(&s2[threadIdx.x],      ls[threadIdx.x]);
        atomicAdd(&s2[16 + threadIdx.x], lss[threadIdx.x]);
    }
}

// ---------------------------------------------------------------------------
// Stage 3: BN2 (folded) + conv3 (16->32, k=5, 15->11) + ReLU + stats (32 ch)
// ---------------------------------------------------------------------------
__global__ __launch_bounds__(256) void conv3_kernel(const float* __restrict__ h2,
                                                    const float* __restrict__ s2,
                                                    const float* __restrict__ g2,
                                                    const float* __restrict__ bb2,
                                                    const float* __restrict__ w,
                                                    const float* __restrict__ b,
                                                    float* __restrict__ h3,
                                                    float* __restrict__ s3) {
    __shared__ float scale[16], shift[16];
    __shared__ float ls[32], lss[32];
    if (threadIdx.x < 16) {
        const float cnt = (float)NROWS * 15.f;
        float m = s2[threadIdx.x] / cnt;
        float v = s2[16 + threadIdx.x] / cnt - m * m;
        float inv = rsqrtf(v + 1e-5f) * g2[threadIdx.x];
        scale[threadIdx.x] = inv;
        shift[threadIdx.x] = bb2[threadIdx.x] - m * inv;
    }
    if (threadIdx.x < 32) { ls[threadIdx.x] = 0.f; lss[threadIdx.x] = 0.f; }
    __syncthreads();
    int id = blockIdx.x * 256 + threadIdx.x;
    const int total = NROWS * 32 * 11;
    if (id < total) {
        int n   = id / (32 * 11);
        int rem = id % (32 * 11);
        int c   = rem / 11;
        int t   = rem % 11;
        float acc = b[c];
        #pragma unroll
        for (int ci = 0; ci < 16; ++ci) {
            float sc = scale[ci], sh = shift[ci];
            const float* hr = h2 + n * 240 + ci * 15;
            const float* wr = w + c * 80 + ci * 5;
            #pragma unroll
            for (int k = 0; k < 5; ++k) acc += (hr[t + k] * sc + sh) * wr[k];
        }
        float val = acc > 0.f ? acc : 0.f;
        h3[id] = val;
        atomicAdd(&ls[c], val);
        atomicAdd(&lss[c], val * val);
    }
    __syncthreads();
    if (threadIdx.x < 32) {
        atomicAdd(&s3[threadIdx.x],      ls[threadIdx.x]);
        atomicAdd(&s3[32 + threadIdx.x], lss[threadIdx.x]);
    }
}

// ---------------------------------------------------------------------------
// Stage 4: BN3 (folded) + linear 352->64 + ReLU, cast to f16 -> eh[n*64+o]
// One 64-thread block per row; normalized activations staged in LDS.
// ---------------------------------------------------------------------------
__global__ __launch_bounds__(64) void linear_kernel(const float* __restrict__ h3,
                                                    const float* __restrict__ s3,
                                                    const float* __restrict__ g3,
                                                    const float* __restrict__ bb3,
                                                    const float* __restrict__ lw,
                                                    const float* __restrict__ lb,
                                                    _Float16* __restrict__ eh) {
    __shared__ float scale[32], shift[32];
    __shared__ float hn[352];
    if (threadIdx.x < 32) {
        const float cnt = (float)NROWS * 11.f;
        float m = s3[threadIdx.x] / cnt;
        float v = s3[32 + threadIdx.x] / cnt - m * m;
        float inv = rsqrtf(v + 1e-5f) * g3[threadIdx.x];
        scale[threadIdx.x] = inv;
        shift[threadIdx.x] = bb3[threadIdx.x] - m * inv;
    }
    __syncthreads();
    int n = blockIdx.x;
    for (int j = threadIdx.x; j < 352; j += 64) {
        int c = j / 11;
        hn[j] = h3[n * 352 + j] * scale[c] + shift[c];
    }
    __syncthreads();
    int o = threadIdx.x;
    float acc = lb[o];
    const float* wr = lw + o * 352;
    for (int j = 0; j < 352; ++j) acc += hn[j] * wr[j];
    acc = acc > 0.f ? acc : 0.f;
    eh[n * 64 + o] = (_Float16)acc;
}

// ---------------------------------------------------------------------------
// Stage 5: A = relu(M1*M2^T - M2*M1^T) via V_WMMA_F32_16X16X32_F16 (K==32).
// e (1 MB f16) is fully L2-resident -> load fragments straight from global.
//
// A-frag (16x32 f16): lane = (r=lane&15, hi=lane>>4);
//   VGPR0..7 = K[16*hi .. 16*hi+15] (contiguous, matches B-frag)
// B-frag (32x16, col n = lane&15): lanes 0-15 K=0..15, lanes 16-31 K=16..31
//   -> one contiguous 16-half load of row (j0+n), offset hi*16.
// D (16x16 f32, 8 VGPR): elem g -> row i0+g+8*hi, col j0+(lane&15).
// ---------------------------------------------------------------------------
__device__ inline v16h load_afrag(const _Float16* base, int i0, int lane) {
    int r = lane & 15, hi = lane >> 4;
    const _Float16* row = base + (size_t)(i0 + r) * 64;
    v8h lo = *(const v8h*)(row + hi * 16);
    v8h hv = *(const v8h*)(row + hi * 16 + 8);
    v16h out;
    #pragma unroll
    for (int q = 0; q < 8; ++q) { out[q] = lo[q]; out[8 + q] = hv[q]; }
    return out;
}

__device__ inline v16h load_bfrag(const _Float16* base, int j0, int lane) {
    int n = lane & 15, hi = lane >> 4;
    const _Float16* row = base + (size_t)(j0 + n) * 64;
    return *(const v16h*)(row + hi * 16);
}

__global__ __launch_bounds__(256) void gemm_antisym_kernel(const _Float16* __restrict__ eh,
                                                           float* __restrict__ A) {
    const int N = NROWS;
    int wave = threadIdx.x >> 5;   // 0..7
    int lane = threadIdx.x & 31;
    int wr = wave & 1;             // 2 wave-rows
    int wc = wave >> 1;            // 4 wave-cols
    int bi = blockIdx.x & 127;     // N/64  row blocks
    int bj = blockIdx.x >> 7;      // N/128 col blocks
    int i0 = bi * 64 + wr * 32;
    int j0 = bj * 128 + wc * 32;

    const _Float16* m1 = eh;       // e[:, :32], row stride 64 halves
    const _Float16* m2 = eh + 32;  // e[:, 32:]

    v16h a1[2], a2[2], b1[2], b2[2];
    #pragma unroll
    for (int t = 0; t < 2; ++t) {
        a1[t] = load_afrag(m1, i0 + t * 16, lane);   // rows of M1 (A operand)
        a2[t] = load_afrag(m2, i0 + t * 16, lane);   // rows of M2 (A operand)
        b1[t] = load_bfrag(m1, j0 + t * 16, lane);   // M1_j^T (B operand)
        b2[t] = load_bfrag(m2, j0 + t * 16, lane);   // M2_j^T (B operand)
    }

    int n = lane & 15, hi = lane >> 4;
    #pragma unroll
    for (int ti = 0; ti < 2; ++ti) {
        #pragma unroll
        for (int tj = 0; tj < 2; ++tj) {
            v8f c1 = {}; v8f c2 = {};
            // D1 = M1_i x M2_j^T   (one WMMA: K = 32 exactly)
            c1 = __builtin_amdgcn_wmma_f32_16x16x32_f16(false, a1[ti], false, b2[tj],
                                                        (short)0, c1, false, false);
            // D2 = M2_i x M1_j^T
            c2 = __builtin_amdgcn_wmma_f32_16x16x32_f16(false, a2[ti], false, b1[tj],
                                                        (short)0, c2, false, false);
            #pragma unroll
            for (int g = 0; g < 8; ++g) {
                float v = c1[g] - c2[g];
                v = v > 0.f ? v : 0.f;
                int row = i0 + ti * 16 + g + 8 * hi;
                int col = j0 + tj * 16 + n;
                A[(size_t)row * N + col] = v;
            }
        }
    }
}

// ---------------------------------------------------------------------------
// Stage 6: per-row top-15 over A (row cached in LDS), emit edges and rewrite
// the row masked in place. One 256-thread block per row.
// ---------------------------------------------------------------------------
__global__ __launch_bounds__(256) void topk_kernel(float* __restrict__ A,
                                                   float* __restrict__ out_src,
                                                   float* __restrict__ out_dst,
                                                   float* __restrict__ out_attr) {
    const int N = NROWS;
    __shared__ float row[NROWS];
    __shared__ float rv[256];
    __shared__ int   ri[256];
    __shared__ float selv[TOPK];
    __shared__ int   seli[TOPK];

    int i = blockIdx.x;
    float* Arow = A + (size_t)i * N;
    for (int j = threadIdx.x; j < N; j += 256) row[j] = Arow[j];
    __syncthreads();

    for (int t = 0; t < TOPK; ++t) {
        float bv = -1e30f; int bidx = 0x7fffffff;
        for (int j = threadIdx.x; j < N; j += 256) {
            float v = row[j];
            if (v > bv || (v == bv && j < bidx)) { bv = v; bidx = j; }
        }
        rv[threadIdx.x] = bv; ri[threadIdx.x] = bidx;
        __syncthreads();
        for (int s = 128; s > 0; s >>= 1) {
            if (threadIdx.x < s) {
                float ov = rv[threadIdx.x + s]; int oi = ri[threadIdx.x + s];
                if (ov > rv[threadIdx.x] ||
                    (ov == rv[threadIdx.x] && oi < ri[threadIdx.x])) {
                    rv[threadIdx.x] = ov; ri[threadIdx.x] = oi;
                }
            }
            __syncthreads();
        }
        if (threadIdx.x == 0) {
            selv[t] = rv[0]; seli[t] = ri[0];
            row[ri[0]] = -1e30f;            // mask out for next pass
        }
        __syncthreads();
    }

    if (threadIdx.x < TOPK) {
        int t = threadIdx.x;
        float v = selv[t]; int idx = seli[t];
        out_src[(size_t)i * TOPK + t]  = (float)i;
        out_dst[(size_t)i * TOPK + t]  = (v > 0.f) ? (float)idx : (float)i;
        out_attr[(size_t)i * TOPK + t] = v;
    }
    __syncthreads();

    // rebuild masked row and write back in place
    for (int j = threadIdx.x; j < N; j += 256) row[j] = 0.f;
    __syncthreads();
    if (threadIdx.x < TOPK) row[seli[threadIdx.x]] = selv[threadIdx.x];
    __syncthreads();
    for (int j = threadIdx.x; j < N; j += 256) Arow[j] = row[j];
}

// ---------------------------------------------------------------------------
// launch
// ---------------------------------------------------------------------------
extern "C" void kernel_launch(void* const* d_in, const int* in_sizes, int n_in,
                              void* d_out, int out_size, void* d_ws, size_t ws_size,
                              hipStream_t stream) {
    const float* x       = (const float*)d_in[0];
    const float* conv1_w = (const float*)d_in[1];
    const float* conv1_b = (const float*)d_in[2];
    const float* bn1_g   = (const float*)d_in[3];
    const float* bn1_b   = (const float*)d_in[4];
    const float* conv2_w = (const float*)d_in[5];
    const float* conv2_b = (const float*)d_in[6];
    const float* bn2_g   = (const float*)d_in[7];
    const float* bn2_b   = (const float*)d_in[8];
    const float* conv3_w = (const float*)d_in[9];
    const float* conv3_b = (const float*)d_in[10];
    const float* bn3_g   = (const float*)d_in[11];
    const float* bn3_b   = (const float*)d_in[12];
    const float* lin_w   = (const float*)d_in[13];
    const float* lin_b   = (const float*)d_in[14];
    (void)in_sizes; (void)n_in; (void)out_size; (void)ws_size;

    const int N = NROWS, K = TOPK;

    // output layout: edge_indices (2*N*K), edge_attr (N*K), A_masked (N*N)
    float* out      = (float*)d_out;
    float* out_src  = out;
    float* out_dst  = out + (size_t)N * K;
    float* out_attr = out + (size_t)2 * N * K;
    float* A        = out + (size_t)3 * N * K;   // also used as raw-A scratch

    // workspace layout
    float* stats = (float*)d_ws;                 // 256 floats (stat accumulators)
    float* s1 = stats;                           // 8 sum + 8 sumsq
    float* s2 = stats + 16;                      // 16 + 16
    float* s3 = stats + 48;                      // 32 + 32
    float* h1 = stats + 256;                     // 8192*152
    float* h2 = h1 + (size_t)N * 152;            // 8192*240
    float* h3 = h2 + (size_t)N * 240;            // 8192*352
    _Float16* eh = (_Float16*)(h3 + (size_t)N * 352); // 8192*64 f16

    hipMemsetAsync(stats, 0, 256 * sizeof(float), stream);

    conv1_kernel<<<(N * 152 + 255) / 256, 256, 0, stream>>>(x, conv1_w, conv1_b, h1, s1);
    conv2_kernel<<<(N * 240 + 255) / 256, 256, 0, stream>>>(h1, s1, bn1_g, bn1_b,
                                                            conv2_w, conv2_b, h2, s2);
    conv3_kernel<<<(N * 352 + 255) / 256, 256, 0, stream>>>(h2, s2, bn2_g, bn2_b,
                                                            conv3_w, conv3_b, h3, s3);
    linear_kernel<<<N, 64, 0, stream>>>(h3, s3, bn3_g, bn3_b, lin_w, lin_b, eh);

    // (N/64) * (N/128) = 128 * 64 = 8192 blocks, 8 waves each, 64 WMMAs/block
    gemm_antisym_kernel<<<(N / 64) * (N / 128), 256, 0, stream>>>(eh, A);

    topk_kernel<<<N, 256, 0, stream>>>(A, out_src, out_dst, out_attr);
}
